// GCNEdgeClassifier_88759794139473
// MI455X (gfx1250) — matrix operand order repaired and checked by reading, not compiled, not run on hardware
//
#include <hip/hip_runtime.h>
#include <math.h>

#define C_  128
#define H_  256
#define NC_ 2
#define NN_ 50000
#define NE_ 1600000
#define NQ_ 500000

typedef __attribute__((ext_vector_type(16))) _Float16 v16h;
typedef __attribute__((ext_vector_type(8)))  float    v8f;

// ---------- WMMA fragment loaders (layouts per CDNA5 ISA) ----------
// A 16x32 f16: lane L -> M = L&15 ; halves 0-7: K = kb..kb+7, halves 8-15: K = kb+16..kb+23
// where kb = kk + 8*(L>=16). Rows contiguous in K -> two 16B chunks.
__device__ __forceinline__ v16h ld_fragA(const _Float16* base, int ld, int kk, int lane) {
  const _Float16* p = base + (size_t)(lane & 15) * ld + kk + ((lane >> 4) << 3);
  v16h r;
  ((float4*)&r)[0] = ((const float4*)p)[0];
  ((float4*)&r)[1] = ((const float4*)(p + 16))[0];
  return r;
}
// B 32x16 f16 stored as W[n][k] row-major: lane L -> N = row + (L&15);
// halves 0-7 -> K = kb..kb+7, halves 8-15 -> K = kb+16..kb+23, kb = kk + 8*(L>=16) (two 16B chunks).
__device__ __forceinline__ v16h ld_fragB(const _Float16* base, int ld, int row, int kk, int lane) {
  const _Float16* p = base + (size_t)(row + (lane & 15)) * ld + kk + ((lane >> 4) << 3);
  v16h r;
  ((float4*)&r)[0] = ((const float4*)p)[0];
  ((float4*)&r)[1] = ((const float4*)(p + 16))[0];
  return r;
}
__device__ __forceinline__ v8f wmma_f16(v16h a, v16h b, v8f c) {
  return __builtin_amdgcn_wmma_f32_16x16x32_f16(false, a, false, b, (short)0, c, false, false);
}

// ---------- small prep kernels ----------
__global__ void fill_kernel(float* p, float v, int n) {
  int i = blockIdx.x * blockDim.x + threadIdx.x;
  if (i < n) p[i] = v;
}

__global__ void prep_kernel(const float* __restrict__ lw0, const float* __restrict__ lw1,
                            const float* g0, const float* b0, const float* m0, const float* v0,
                            const float* g1, const float* b1, const float* m1, const float* v1,
                            _Float16* lw0h, _Float16* lw1h, float* bn) {
  int i = blockIdx.x * blockDim.x + threadIdx.x;
  const int tot = H_ * C_ + H_ * H_;
  for (int idx = i; idx < tot; idx += gridDim.x * blockDim.x) {
    if (idx < H_ * C_) lw0h[idx] = (_Float16)lw0[idx];
    else               lw1h[idx - H_ * C_] = (_Float16)lw1[idx - H_ * C_];
  }
  if (i < H_) {
    float s0 = g0[i] * rsqrtf(v0[i] + 1e-5f);
    bn[i]          = s0;
    bn[H_ + i]     = b0[i] - m0[i] * s0;
    float s1 = g1[i] * rsqrtf(v1[i] + 1e-5f);
    bn[2 * H_ + i] = s1;
    bn[3 * H_ + i] = b1[i] - m1[i] * s1;
  }
}

// ---------- GRU weight evolution: w = GRUCell(w_init, w_init); emit w^T in f16 ----------
__global__ __launch_bounds__(C_) void gru_kernel(const float* __restrict__ w_init,
                                                 const float* __restrict__ wih,
                                                 const float* __restrict__ whh,
                                                 const float* __restrict__ bih,
                                                 const float* __restrict__ bhh,
                                                 _Float16* __restrict__ wt) {
  int i = blockIdx.x;    // row of w_init (batch row)
  int j = threadIdx.x;   // output column
  __shared__ float row[C_];
  row[j] = w_init[i * C_ + j];
  __syncthreads();
  float gi0 = bih[j], gi1 = bih[C_ + j], gi2 = bih[2 * C_ + j];
  float gh0 = bhh[j], gh1 = bhh[C_ + j], gh2 = bhh[2 * C_ + j];
  const float* wr0 = wih + (size_t)(0 * C_ + j) * C_;
  const float* wr1 = wih + (size_t)(1 * C_ + j) * C_;
  const float* wr2 = wih + (size_t)(2 * C_ + j) * C_;
  const float* hr0 = whh + (size_t)(0 * C_ + j) * C_;
  const float* hr1 = whh + (size_t)(1 * C_ + j) * C_;
  const float* hr2 = whh + (size_t)(2 * C_ + j) * C_;
  for (int k = 0; k < C_; ++k) {
    float xv = row[k];
    gi0 += xv * wr0[k]; gi1 += xv * wr1[k]; gi2 += xv * wr2[k];
    gh0 += xv * hr0[k]; gh1 += xv * hr1[k]; gh2 += xv * hr2[k];
  }
  float r = 1.0f / (1.0f + expf(-(gi0 + gh0)));
  float z = 1.0f / (1.0f + expf(-(gi1 + gh1)));
  float n = tanhf(gi2 + r * gh2);
  float w = (1.0f - z) * n + z * row[j];
  wt[(size_t)j * C_ + i] = (_Float16)w;   // transposed: wt[out][k]
}

// ---------- degree / dinv ----------
__global__ void deg_kernel(const int* __restrict__ dst, const float* __restrict__ ew,
                           float* __restrict__ deg, int E) {
  int e = blockIdx.x * blockDim.x + threadIdx.x;
  if (e < E) atomicAdd(&deg[dst[e]], ew[e]);
}
__global__ void dinv_kernel(float* d, int n) {
  int i = blockIdx.x * blockDim.x + threadIdx.x;
  if (i < n) d[i] = rsqrtf(d[i] + 1.0f);
}

// ---------- xw = (relu?)(x) @ w   (WMMA; weights staged in LDS) ----------
__global__ __launch_bounds__(128) void gemm_xw_kernel(const float* __restrict__ xin,
                                                      const _Float16* __restrict__ wt,
                                                      float* __restrict__ xw,
                                                      int nrows, int apply_relu) {
  __shared__ _Float16 atile[4][16 * C_];   // 16 KB, one strip per wave
  __shared__ _Float16 wlds[C_ * C_];       // 32 KB weight stage
  int tid = threadIdx.x, lane = tid & 31, wave = tid >> 5;
  int rbase = blockIdx.x * 64 + wave * 16;

  // cooperative weight stage: 32 KB as float4 chunks
  {
    const float4* s4 = (const float4*)wt;
    float4* d4 = (float4*)wlds;
    #pragma unroll
    for (int i = 0; i < (C_ * C_ / 8) / 128; ++i)
      d4[tid + i * 128] = s4[tid + i * 128];
  }

  // A strip (wave-private)
  for (int rr = 0; rr < 16; ++rr) {
    int row = rbase + rr;
    float4 v = make_float4(0.f, 0.f, 0.f, 0.f);
    if (row < nrows) v = ((const float4*)(xin + (size_t)row * C_))[lane];
    if (apply_relu) {
      v.x = fmaxf(v.x, 0.f); v.y = fmaxf(v.y, 0.f);
      v.z = fmaxf(v.z, 0.f); v.w = fmaxf(v.w, 0.f);
    }
    _Float16* d = &atile[wave][rr * C_ + lane * 4];
    d[0] = (_Float16)v.x; d[1] = (_Float16)v.y; d[2] = (_Float16)v.z; d[3] = (_Float16)v.w;
  }
  __syncthreads();

  // preload invariant A fragments
  v16h a0 = ld_fragA(atile[wave], C_, 0,  lane);
  v16h a1 = ld_fragA(atile[wave], C_, 32, lane);
  v16h a2 = ld_fragA(atile[wave], C_, 64, lane);
  v16h a3 = ld_fragA(atile[wave], C_, 96, lane);

  bool full = (rbase + 16 <= nrows);
  for (int nt = 0; nt < 8; ++nt) {
    v16h b0 = ld_fragB(wlds, C_, nt * 16, 0,  lane);
    v16h b1 = ld_fragB(wlds, C_, nt * 16, 32, lane);
    v16h b2 = ld_fragB(wlds, C_, nt * 16, 64, lane);
    v16h b3 = ld_fragB(wlds, C_, nt * 16, 96, lane);
    v8f acc = {};
    acc = wmma_f16(a0, b0, acc);
    acc = wmma_f16(a1, b1, acc);
    acc = wmma_f16(a2, b2, acc);
    acc = wmma_f16(a3, b3, acc);
    int col = nt * 16 + (lane & 15);
    int ro  = (lane >> 4) << 3;
    float* op = xw + (size_t)(rbase + ro) * C_ + col;
    if (full) {
      #pragma unroll
      for (int v = 0; v < 8; ++v) op[(size_t)v * C_] = acc[v];
    } else {
      for (int v = 0; v < 8; ++v)
        if (rbase + ro + v < nrows) op[(size_t)v * C_] = acc[v];
    }
  }
}

// ---------- self-loop init: agg = xw * dinv^2 ----------
__global__ void selfloop_kernel(const float* __restrict__ xw, const float* __restrict__ dinv,
                                float* __restrict__ agg, int nvec) {
  int i = blockIdx.x * blockDim.x + threadIdx.x;
  if (i < nvec) {
    int node = i / (C_ / 4);
    float d = dinv[node];
    float s = d * d;
    float4 v = ((const float4*)xw)[i];
    v.x *= s; v.y *= s; v.z *= s; v.w *= s;
    ((float4*)agg)[i] = v;
  }
}

// ---------- edge scatter: agg[dst] += xw[src] * norm (L2-resident f32 atomics) ----------
__global__ __launch_bounds__(256) void scatter_kernel(const int* __restrict__ src,
                                                      const int* __restrict__ dst,
                                                      const float* __restrict__ ew,
                                                      const float* __restrict__ dinv,
                                                      const float* __restrict__ xw,
                                                      float* __restrict__ agg, int E) {
  int e = blockIdx.x * 8 + (threadIdx.x >> 5);
  if (e >= E) return;
  int lane = threadIdx.x & 31;
  int s = src[e], d = dst[e];
  float norm = dinv[s] * ew[e] * dinv[d];
  float4 v = ((const float4*)(xw + (size_t)s * C_))[lane];
  float* out = agg + (size_t)d * C_ + lane * 4;
  atomicAdd(out + 0, v.x * norm);
  atomicAdd(out + 1, v.y * norm);
  atomicAdd(out + 2, v.z * norm);
  atomicAdd(out + 3, v.w * norm);
}

// ---------- fused edge-MLP head: gather*gather -> FC+BN+ReLU x2 -> proj -> log_softmax ----------
__global__ __launch_bounds__(128) void mlp_kernel(const float* __restrict__ x,
                                                  const int* __restrict__ e1,
                                                  const int* __restrict__ e2,
                                                  const _Float16* __restrict__ w0h,
                                                  const _Float16* __restrict__ w1h,
                                                  const float* __restrict__ bn,
                                                  const float* __restrict__ b0,
                                                  const float* __restrict__ b1,
                                                  const float* __restrict__ w2,
                                                  const float* __restrict__ b2,
                                                  float* __restrict__ out, int Q) {
  __shared__ _Float16 h0[4][16 * C_];   // 16 KB
  __shared__ _Float16 h1[4][16 * H_];   // 32 KB
  int lane = threadIdx.x & 31, wave = threadIdx.x >> 5;
  int rbase = blockIdx.x * 64 + wave * 16;
  __builtin_prefetch(w0h, 0, 1);
  __builtin_prefetch(w1h, 0, 1);

  // gather + hadamard with fused relu on both operands
  for (int rr = 0; rr < 16; ++rr) {
    int q = rbase + rr;
    float4 p = make_float4(0.f, 0.f, 0.f, 0.f);
    if (q < Q) {
      int a = e1[q], b = e2[q];
      float4 va = ((const float4*)(x + (size_t)a * C_))[lane];
      float4 vb = ((const float4*)(x + (size_t)b * C_))[lane];
      p.x = fmaxf(va.x, 0.f) * fmaxf(vb.x, 0.f);
      p.y = fmaxf(va.y, 0.f) * fmaxf(vb.y, 0.f);
      p.z = fmaxf(va.z, 0.f) * fmaxf(vb.z, 0.f);
      p.w = fmaxf(va.w, 0.f) * fmaxf(vb.w, 0.f);
    }
    _Float16* dp = &h0[wave][rr * C_ + lane * 4];
    dp[0] = (_Float16)p.x; dp[1] = (_Float16)p.y; dp[2] = (_Float16)p.z; dp[3] = (_Float16)p.w;
  }

  // layer 1: h1 = relu( (h0 @ W0^T + b0) * bnS0 + bnB0 )   K=128, N=256
  {
    v16h a0 = ld_fragA(h0[wave], C_, 0,  lane);
    v16h a1 = ld_fragA(h0[wave], C_, 32, lane);
    v16h a2 = ld_fragA(h0[wave], C_, 64, lane);
    v16h a3 = ld_fragA(h0[wave], C_, 96, lane);
    for (int nt = 0; nt < 16; ++nt) {
      v16h b0f = ld_fragB(w0h, C_, nt * 16, 0,  lane);
      v16h b1f = ld_fragB(w0h, C_, nt * 16, 32, lane);
      v16h b2f = ld_fragB(w0h, C_, nt * 16, 64, lane);
      v16h b3f = ld_fragB(w0h, C_, nt * 16, 96, lane);
      v8f acc = {};
      acc = wmma_f16(a0, b0f, acc);
      acc = wmma_f16(a1, b1f, acc);
      acc = wmma_f16(a2, b2f, acc);
      acc = wmma_f16(a3, b3f, acc);
      int col = nt * 16 + (lane & 15);
      float s = bn[col];
      float t = bn[H_ + col] + b0[col] * s;
      int ro = (lane >> 4) << 3;
      #pragma unroll
      for (int v = 0; v < 8; ++v)
        h1[wave][(ro + v) * H_ + col] = (_Float16)fmaxf(acc[v] * s + t, 0.0f);
    }
  }

  // layer 2 (K=256, N=256) fused with layer-3 projection to NC=2 in registers
  float lacc0[8] = {}, lacc1[8] = {};
  {
    v16h a[8];
    #pragma unroll
    for (int k = 0; k < 8; ++k) a[k] = ld_fragA(h1[wave], H_, k * 32, lane);
    for (int nt = 0; nt < 16; ++nt) {
      v16h bfr[8];
      #pragma unroll
      for (int k = 0; k < 8; ++k) bfr[k] = ld_fragB(w1h, H_, nt * 16, k * 32, lane);
      v8f acc = {};
      #pragma unroll
      for (int k = 0; k < 8; ++k) acc = wmma_f16(a[k], bfr[k], acc);
      int col = nt * 16 + (lane & 15);
      float s = bn[2 * H_ + col];
      float t = bn[3 * H_ + col] + b1[col] * s;
      float w20 = w2[col];        // lin_w2[0][col]
      float w21 = w2[H_ + col];   // lin_w2[1][col]
      #pragma unroll
      for (int v = 0; v < 8; ++v) {
        float h2v = fmaxf(acc[v] * s + t, 0.0f);
        lacc0[v] += h2v * w20;
        lacc1[v] += h2v * w21;
      }
    }
  }
  // reduce across the 16 lanes sharing the same row set (xor within 16-lane groups)
  for (int m = 1; m <= 8; m <<= 1)
    for (int v = 0; v < 8; ++v) {
      lacc0[v] += __shfl_xor(lacc0[v], m, 32);
      lacc1[v] += __shfl_xor(lacc1[v], m, 32);
    }
  if ((lane & 15) == 0) {
    int ro = (lane >> 4) << 3;
    float bb0 = b2[0], bb1 = b2[1];
    for (int v = 0; v < 8; ++v) {
      int q = rbase + ro + v;
      if (q < Q) {
        float l0 = lacc0[v] + bb0, l1 = lacc1[v] + bb1;
        float mx = fmaxf(l0, l1);
        float lse = mx + logf(expf(l0 - mx) + expf(l1 - mx));
        out[(size_t)q * 2 + 0] = l0 - lse;
        out[(size_t)q * 2 + 1] = l1 - lse;
      }
    }
  }
}

extern "C" void kernel_launch(void* const* d_in, const int* in_sizes, int n_in,
                              void* d_out, int out_size, void* d_ws, size_t ws_size,
                              hipStream_t stream) {
  (void)in_sizes; (void)n_in; (void)out_size; (void)ws_size;
  const float* X  = (const float*)d_in[0];
  const int*   ei = (const int*)d_in[1];
  const float* ew = (const float*)d_in[2];
  const int*   e1 = (const int*)d_in[3];
  const int*   e2 = (const int*)d_in[4];
  const float* w_init[2] = { (const float*)d_in[5],  (const float*)d_in[10] };
  const float* w_ih[2]   = { (const float*)d_in[6],  (const float*)d_in[11] };
  const float* w_hh[2]   = { (const float*)d_in[7],  (const float*)d_in[12] };
  const float* b_ih[2]   = { (const float*)d_in[8],  (const float*)d_in[13] };
  const float* b_hh[2]   = { (const float*)d_in[9],  (const float*)d_in[14] };
  const float* lin_w0 = (const float*)d_in[15];
  const float* lin_b0 = (const float*)d_in[16];
  const float* lin_w1 = (const float*)d_in[17];
  const float* lin_b1 = (const float*)d_in[18];
  const float* lin_w2 = (const float*)d_in[19];
  const float* lin_b2 = (const float*)d_in[20];
  const float* bn_g0 = (const float*)d_in[21];
  const float* bn_b0 = (const float*)d_in[22];
  const float* bn_m0 = (const float*)d_in[23];
  const float* bn_v0 = (const float*)d_in[24];
  const float* bn_g1 = (const float*)d_in[25];
  const float* bn_b1 = (const float*)d_in[26];
  const float* bn_m1 = (const float*)d_in[27];
  const float* bn_v1 = (const float*)d_in[28];

  // workspace carve-out (256B aligned)
  char* ws = (char*)d_ws;
  size_t off = 0;
  auto alloc = [&](size_t bytes) -> char* {
    char* p = ws + off;
    off += (bytes + 255) & ~(size_t)255;
    return p;
  };
  float*    dinv  = (float*)   alloc((size_t)NN_ * 4);
  _Float16* wt0   = (_Float16*)alloc((size_t)C_ * C_ * 2);
  _Float16* wt1   = (_Float16*)alloc((size_t)C_ * C_ * 2);
  _Float16* lw0h  = (_Float16*)alloc((size_t)H_ * C_ * 2);
  _Float16* lw1h  = (_Float16*)alloc((size_t)H_ * H_ * 2);
  float*    bnbuf = (float*)   alloc((size_t)4 * H_ * 4);
  float*    xw    = (float*)   alloc((size_t)NN_ * C_ * 4);
  float*    agg   = (float*)   alloc((size_t)NN_ * C_ * 4);

  const int* src = ei;
  const int* dst = ei + NE_;
  float* outp = (float*)d_out;

  // prep: f16 weights + folded BN
  prep_kernel<<<(H_ * H_ + H_ * C_ + 255) / 256, 256, 0, stream>>>(
      lin_w0, lin_w1, bn_g0, bn_b0, bn_m0, bn_v0, bn_g1, bn_b1, bn_m1, bn_v1,
      lw0h, lw1h, bnbuf);
  // evolve GCN weights
  gru_kernel<<<C_, C_, 0, stream>>>(w_init[0], w_ih[0], w_hh[0], b_ih[0], b_hh[0], wt0);
  gru_kernel<<<C_, C_, 0, stream>>>(w_init[1], w_ih[1], w_hh[1], b_ih[1], b_hh[1], wt1);
  // degree -> dinv (shared by both layers)
  fill_kernel<<<(NN_ + 255) / 256, 256, 0, stream>>>(dinv, 0.0f, NN_);
  deg_kernel<<<(NE_ + 255) / 256, 256, 0, stream>>>(dst, ew, dinv, NE_);
  dinv_kernel<<<(NN_ + 255) / 256, 256, 0, stream>>>(dinv, NN_);

  const int gemm_grid = (NN_ + 63) / 64;
  const int sl_grid   = (NN_ * (C_ / 4) + 255) / 256;
  const int sc_grid   = (NE_ + 7) / 8;

  // GCN layer 0: xw = X @ w0 ; agg = self-loop + edge scatter
  gemm_xw_kernel<<<gemm_grid, 128, 0, stream>>>(X, wt0, xw, NN_, 0);
  selfloop_kernel<<<sl_grid, 256, 0, stream>>>(xw, dinv, agg, NN_ * (C_ / 4));
  scatter_kernel<<<sc_grid, 256, 0, stream>>>(src, dst, ew, dinv, xw, agg, NE_);

  // GCN layer 1: xw = relu(agg) @ w1 ; agg = self-loop + edge scatter
  gemm_xw_kernel<<<gemm_grid, 128, 0, stream>>>(agg, wt1, xw, NN_, 1);
  selfloop_kernel<<<sl_grid, 256, 0, stream>>>(xw, dinv, agg, NN_ * (C_ / 4));
  scatter_kernel<<<sc_grid, 256, 0, stream>>>(src, dst, ew, dinv, xw, agg, NE_);

  // fused edge-MLP head (relu on gather)
  mlp_kernel<<<(NQ_ + 63) / 64, 128, 0, stream>>>(
      agg, e1, e2, lw0h, lw1h, bnbuf, lin_b0, lin_b1, lin_w2, lin_b2, outp, NQ_);
}
